// VigResnet_60507499266078
// MI455X (gfx1250) — matrix operand order repaired and checked by reading, not compiled, not run on hardware
//
#include <hip/hip_runtime.h>
#include <hip/hip_bf16.h>
#include <math.h>

// ---------------------------------------------------------------------------
// Problem constants (from reference): B=256, N=196, D=2048, F=512, k=5
// ---------------------------------------------------------------------------
#define BATCH 256
#define NPTS  196
#define DIM   2048
#define FDIM  512
#define KNN   5
#define BN    (BATCH * NPTS)     // 50176 rows
#define NPAD  208                // 13 * 16  (N padded to 16-multiple)
#define NTILE 13                 // column tiles of 16
#define KC    32                 // K-chunk = WMMA K for bf16
#define LSTR  36                 // LDS row stride in bf16 elements (bank pad)
#define LEAKY 0.2f

typedef __attribute__((ext_vector_type(16))) __bf16 v16bf;
typedef __attribute__((ext_vector_type(8)))  float  v8f;

__device__ __forceinline__ v8f wmma_bf16(v16bf a, v16bf b, v8f c) {
  // (neg_a, A, neg_b, B, c_mod, C, reuse_a, reuse_b)
  return __builtin_amdgcn_wmma_f32_16x16x32_bf16(false, a, false, b,
                                                 (short)0, c, false, false);
}

// A-matrix fragment (16x32 bf16) from row-major LDS tile [16][K].
// ISA 7.12.2: lanes 0-15 -> M=lane, lanes 16-31 same M, upper K groups.
// VGPR v (pair p): K = (v<4) ? 8*h + 2v + p : 16 + 8*h + 2(v-4) + p.
__device__ __forceinline__ v16bf frag_a_rows(const __bf16* tile, int stride, int lane) {
  const int m  = lane & 15;
  const int hh = (lane >> 4) & 1;
  const __bf16* r = tile + m * stride;
  v16bf f;
#pragma unroll
  for (int v = 0; v < 8; ++v) {
    const int kb = (v < 4) ? (8 * hh + 2 * v) : (16 + 8 * hh + 2 * (v - 4));
    f[2 * v]     = r[kb];
    f[2 * v + 1] = r[kb + 1];
  }
  return f;
}

// B-matrix fragment (32x16 bf16) where element (k,n) is stored row-major as
// storage[n][k] (i.e. B = storedᵀ: natural for Gram / W-transposed LDS tiles).
// ISA: lane n = lane&15 selects B column; K = (v<4) ? 8*h + 2v + p : 16 + 8*h + 2(v-4) + p.
__device__ __forceinline__ v16bf frag_b_cols(const __bf16* tile, int stride, int lane) {
  const int n  = lane & 15;
  const int hh = (lane >> 4) & 1;
  const __bf16* r = tile + n * stride;
  v16bf f;
#pragma unroll
  for (int v = 0; v < 8; ++v) {
    const int kb = (v < 4) ? (8 * hh + 2 * v) : (16 + 8 * hh + 2 * (v - 4));
    f[2 * v]     = r[kb];
    f[2 * v + 1] = r[kb + 1];
  }
  return f;
}

// ---------------------------------------------------------------------------
// Kernel 1: row squared norms  sq[b*N+n] = sum_d x^2
// ---------------------------------------------------------------------------
__global__ __launch_bounds__(256) void norms_kernel(const float* __restrict__ x,
                                                    float* __restrict__ sq) {
  __shared__ float red[256];
  const size_t row = blockIdx.x;
  const int tid = threadIdx.x;
  const float* xr = x + row * (size_t)DIM;
  float s = 0.f;
#pragma unroll 4
  for (int f = tid; f < DIM; f += 256) {
    float v = xr[f];
    s += v * v;
  }
  red[tid] = s;
  __syncthreads();
  for (int off = 128; off > 0; off >>= 1) {
    if (tid < off) red[tid] += red[tid + off];
    __syncthreads();
  }
  if (tid == 0) sq[row] = red[0];
}

// ---------------------------------------------------------------------------
// Kernel 2: per-batch Gram + fused top-5 selection.
// One workgroup per (batch, PAIR of 16-row tiles): grid (256, 7).
// 4 waves; wave w owns column tiles j = w + 4t, t=0..3, with j CLAMPED to 12
// so the WMMA loop is branch-free (EXEC always all-1s); clamped duplicates
// are discarded at the score-write stage (scalar-uniform branches only).
// x is split hi/lo bf16 so G ~= hi*hi + hi*lo + lo*hi recovers ~fp32 accuracy.
// Ranking score for -d2: s = 2*G[i,j] - sq[j] (row-constant -sq[i] dropped).
// ---------------------------------------------------------------------------
__global__ __launch_bounds__(128) void gram_topk_kernel(const float* __restrict__ x,
                                                        const float* __restrict__ sq,
                                                        int* __restrict__ nbr) {
  __shared__ __bf16 sHi[NPAD * LSTR];
  __shared__ __bf16 sLo[NPAD * LSTR];
  __shared__ float  sScore[32 * NPAD];

  const int b    = blockIdx.x;
  const int rt0  = blockIdx.y * 2;                 // first row tile (0,2,..,12)
  const int rt1  = rt0 + 1;                        // may be 13 (invalid)
  const int rt1c = (rt1 < NTILE) ? rt1 : (NTILE - 1);
  const int tid  = threadIdx.x;
  const int lane = tid & 31;
  const int wave = __builtin_amdgcn_readfirstlane(tid >> 5);  // scalar-uniform

  const float* xb = x + (size_t)b * NPTS * DIM;

  const v8f vzero = {0.f, 0.f, 0.f, 0.f, 0.f, 0.f, 0.f, 0.f};
  v8f acc[2][4];
#pragma unroll
  for (int r = 0; r < 2; ++r)
#pragma unroll
    for (int t = 0; t < 4; ++t) acc[r][t] = vzero;

  for (int k0 = 0; k0 < DIM; k0 += KC) {
    // Stage all 208 padded rows of this batch for this K-chunk, split hi/lo.
    for (int idx = tid; idx < NPAD * KC; idx += 128) {
      const int r  = idx >> 5;
      const int kk = idx & 31;
      float v = 0.f;
      if (r < NPTS) {
        v = xb[(size_t)r * DIM + k0 + kk];
        if (k0 + KC < DIM)
          __builtin_prefetch(&xb[(size_t)r * DIM + k0 + KC + kk], 0, 1);
      }
      const __bf16 hi = (__bf16)v;
      const __bf16 lo = (__bf16)(v - (float)hi);
      sHi[r * LSTR + kk] = hi;
      sLo[r * LSTR + kk] = lo;
    }
    __syncthreads();

    const v16bf ah0 = frag_a_rows(sHi + rt0  * 16 * LSTR, LSTR, lane);
    const v16bf al0 = frag_a_rows(sLo + rt0  * 16 * LSTR, LSTR, lane);
    const v16bf ah1 = frag_a_rows(sHi + rt1c * 16 * LSTR, LSTR, lane);
    const v16bf al1 = frag_a_rows(sLo + rt1c * 16 * LSTR, LSTR, lane);
#pragma unroll
    for (int t = 0; t < 4; ++t) {
      const int j  = wave + 4 * t;
      const int jc = (j < NTILE) ? j : (NTILE - 1);   // clamp: branch-free
      const v16bf bh = frag_b_cols(sHi + jc * 16 * LSTR, LSTR, lane);
      const v16bf bl = frag_b_cols(sLo + jc * 16 * LSTR, LSTR, lane);
      acc[0][t] = wmma_bf16(ah0, bh, acc[0][t]);
      acc[0][t] = wmma_bf16(ah0, bl, acc[0][t]);
      acc[0][t] = wmma_bf16(al0, bh, acc[0][t]);
      acc[1][t] = wmma_bf16(ah1, bh, acc[1][t]);
      acc[1][t] = wmma_bf16(ah1, bl, acc[1][t]);
      acc[1][t] = wmma_bf16(al1, bh, acc[1][t]);
    }
    __syncthreads();
  }

  // Emit ranking scores to LDS. C/D layout: VGPR i, lane -> M = i + 8*(lane>>4),
  // column = lane & 15. Guards below are scalar-uniform (wave is readfirstlane).
  const float* sqb = sq + (size_t)b * NPTS;
  const int n0 = lane & 15;
  const int hh = lane >> 4;
#pragma unroll
  for (int t = 0; t < 4; ++t) {
    const int j = wave + 4 * t;
    if (j < NTILE) {
      const int col = j * 16 + n0;
      const float sqj = (col < NPTS) ? sqb[col] : 0.f;
      const float pen = (col < NPTS) ? 0.f : -3.0e38f;
#pragma unroll
      for (int i = 0; i < 8; ++i) {
        const int m = i + 8 * hh;
        sScore[m * NPAD + col] = 2.f * acc[0][t][i] - sqj + pen;
      }
      if (rt1 < NTILE) {
#pragma unroll
        for (int i = 0; i < 8; ++i) {
          const int m = 16 + i + 8 * hh;
          sScore[m * NPAD + col] = 2.f * acc[1][t][i] - sqj + pen;
        }
      }
    }
  }
  __syncthreads();

  // Top-5 per row: 32 rows handled by threads 0..31 (serial scan of 208, tiny).
  if (tid < 32) {
    const int row = rt0 * 16 + tid;
    if (row < NPTS) {
      float best[KNN];
      int   bidx[KNN];
#pragma unroll
      for (int u = 0; u < KNN; ++u) { best[u] = -3.4e38f; bidx[u] = 0; }
      const float* sr = sScore + tid * NPAD;
      for (int c = 0; c < NPAD; ++c) {
        const float v = sr[c];
        if (v > best[KNN - 1]) {
          int p = KNN - 1;
          while (p > 0 && v > best[p - 1]) {
            best[p] = best[p - 1];
            bidx[p] = bidx[p - 1];
            --p;
          }
          best[p] = v;
          bidx[p] = c;
        }
      }
      int* o = nbr + ((size_t)b * NPTS + row) * KNN;
#pragma unroll
      for (int u = 0; u < KNN; ++u) o[u] = bidx[u];
    }
  }
}

// ---------------------------------------------------------------------------
// Kernel 3: h = x @ W (bf16 WMMA, fp32 accumulate).
// Grid: (BN/128, FDIM/256). 256 threads = 8 waves in 2x4 layout; each wave
// computes a 64x64 block = 4x4 WMMA tiles (16 WMMA per K-chunk vs 8 fragment
// builds). No branches anywhere near WMMA.
// ---------------------------------------------------------------------------
__global__ __launch_bounds__(256) void gemm_h_kernel(const float* __restrict__ x,
                                                     const float* __restrict__ W,
                                                     float* __restrict__ h) {
  __shared__ __bf16 sA[128 * LSTR];    // [row][k]
  __shared__ __bf16 sB[256 * LSTR];    // [col][k]  (W transposed into LDS)

  const int tid  = threadIdx.x;
  const int lane = tid & 31;
  const int wave = __builtin_amdgcn_readfirstlane(tid >> 5);
  const int wr   = wave >> 2;          // 0..1
  const int wc   = wave & 3;           // 0..3
  const size_t rowbase = (size_t)blockIdx.x * 128;
  const int colbase    = blockIdx.y * 256;

  const v8f vzero = {0.f, 0.f, 0.f, 0.f, 0.f, 0.f, 0.f, 0.f};
  v8f acc[4][4];
#pragma unroll
  for (int s = 0; s < 4; ++s)
#pragma unroll
    for (int t = 0; t < 4; ++t) acc[s][t] = vzero;

  for (int k0 = 0; k0 < DIM; k0 += KC) {
    // Stage A (128 x 32): contiguous-k global reads, coalesced.
    for (int idx = tid; idx < 128 * KC; idx += 256) {
      const int r  = idx >> 5;
      const int kk = idx & 31;
      const size_t ga = (rowbase + r) * (size_t)DIM + k0 + kk;
      sA[r * LSTR + kk] = (__bf16)x[ga];
      if (k0 + KC < DIM) __builtin_prefetch(&x[ga + KC], 0, 1);
    }
    // Stage B = W[k0:k0+32, colbase:colbase+256] transposed into [col][k]:
    // consecutive tid walk consecutive columns (contiguous in W) -> coalesced.
    for (int idx = tid; idx < 256 * KC; idx += 256) {
      const int kk = idx >> 8;         // 0..31
      const int c  = idx & 255;        // 0..255
      sB[c * LSTR + kk] = (__bf16)W[(size_t)(k0 + kk) * FDIM + colbase + c];
    }
    __syncthreads();

    v16bf a[4], bfr[4];
#pragma unroll
    for (int s = 0; s < 4; ++s)
      a[s] = frag_a_rows(sA + (wr * 64 + s * 16) * LSTR, LSTR, lane);
#pragma unroll
    for (int t = 0; t < 4; ++t)
      bfr[t] = frag_b_cols(sB + (wc * 64 + t * 16) * LSTR, LSTR, lane);
#pragma unroll
    for (int s = 0; s < 4; ++s)
#pragma unroll
      for (int t = 0; t < 4; ++t)
        acc[s][t] = wmma_bf16(a[s], bfr[t], acc[s][t]);
    __syncthreads();
  }

  const int n0 = lane & 15;
  const int hh = lane >> 4;
#pragma unroll
  for (int s = 0; s < 4; ++s)
#pragma unroll
    for (int t = 0; t < 4; ++t)
#pragma unroll
      for (int i = 0; i < 8; ++i) {
        const size_t row = rowbase + wr * 64 + s * 16 + i + 8 * hh;
        const int    col = colbase + wc * 64 + t * 16 + n0;
        h[row * FDIM + col] = acc[s][t][i];
      }
}

// ---------------------------------------------------------------------------
// Kernel 4: e_src/e_dst = h . a_src / a_dst
// ---------------------------------------------------------------------------
__global__ __launch_bounds__(128) void edge_kernel(const float* __restrict__ h,
                                                   const float* __restrict__ a_src,
                                                   const float* __restrict__ a_dst,
                                                   float* __restrict__ esrc,
                                                   float* __restrict__ edst) {
  __shared__ float rs[128];
  __shared__ float rd[128];
  const size_t row = blockIdx.x;
  const int tid = threadIdx.x;
  const float* hr = h + row * (size_t)FDIM;
  float ss = 0.f, sd = 0.f;
#pragma unroll
  for (int f = tid; f < FDIM; f += 128) {
    const float v = hr[f];
    ss += v * a_src[f];
    sd += v * a_dst[f];
  }
  rs[tid] = ss; rd[tid] = sd;
  __syncthreads();
  for (int off = 64; off > 0; off >>= 1) {
    if (tid < off) { rs[tid] += rs[tid + off]; rd[tid] += rd[tid + off]; }
    __syncthreads();
  }
  if (tid == 0) { esrc[row] = rs[0]; edst[row] = rd[0]; }
}

// ---------------------------------------------------------------------------
// Kernel 5: sparse attention aggregation + ELU.
// softmax over only the 5 neighbors (NEG_INF entries underflow to 0 in fp32).
// ---------------------------------------------------------------------------
__global__ __launch_bounds__(128) void gather_kernel(const float* __restrict__ h,
                                                     const float* __restrict__ esrc,
                                                     const float* __restrict__ edst,
                                                     const int* __restrict__ nbr,
                                                     float* __restrict__ out) {
  __shared__ float sAlpha[KNN];
  __shared__ int   sIdx[KNN];
  const size_t row = blockIdx.x;
  const int tid = threadIdx.x;
  const int b = (int)(row / NPTS);

  if (tid == 0) {
    const int* nb = nbr + row * KNN;
    const float es = esrc[row];
    float e[KNN];
    float mx = -3.4e38f;
#pragma unroll
    for (int j = 0; j < KNN; ++j) {
      const int g = b * NPTS + nb[j];
      float v = es + edst[g];
      v = (v > 0.f) ? v : LEAKY * v;     // leaky_relu before masking
      e[j] = v;
      mx = fmaxf(mx, v);
      sIdx[j] = g;
    }
    float denom = 0.f;
#pragma unroll
    for (int j = 0; j < KNN; ++j) { e[j] = __expf(e[j] - mx); denom += e[j]; }
    const float inv = 1.f / denom;
#pragma unroll
    for (int j = 0; j < KNN; ++j) sAlpha[j] = e[j] * inv;
  }
  __syncthreads();

  float alpha[KNN];
  size_t base[KNN];
#pragma unroll
  for (int j = 0; j < KNN; ++j) {
    alpha[j] = sAlpha[j];
    base[j]  = (size_t)sIdx[j] * FDIM;
  }
#pragma unroll
  for (int f = tid; f < FDIM; f += 128) {
    float s = 0.f;
#pragma unroll
    for (int j = 0; j < KNN; ++j) s += alpha[j] * h[base[j] + f];
    out[row * FDIM + f] = (s > 0.f) ? s : (__expf(s) - 1.f);
  }
}

// ---------------------------------------------------------------------------
// Launch
// ---------------------------------------------------------------------------
extern "C" void kernel_launch(void* const* d_in, const int* in_sizes, int n_in,
                              void* d_out, int out_size, void* d_ws, size_t ws_size,
                              hipStream_t stream) {
  const float* x     = (const float*)d_in[0];   // [B,N,D]
  const float* W     = (const float*)d_in[1];   // [D,F]
  const float* a_src = (const float*)d_in[2];   // [F]
  const float* a_dst = (const float*)d_in[3];   // [F]
  float* out = (float*)d_out;                   // [B,N,F]

  char* ws = (char*)d_ws;
  size_t off = 0;
  auto alloc = [&](size_t bytes) {
    void* p = ws + off;
    off = (off + bytes + 255) & ~(size_t)255;
    return p;
  };
  float* h    = (float*)alloc((size_t)BN * FDIM * sizeof(float));  // ~103 MB
  float* sq   = (float*)alloc((size_t)BN * sizeof(float));
  float* esrc = (float*)alloc((size_t)BN * sizeof(float));
  float* edst = (float*)alloc((size_t)BN * sizeof(float));
  int*   nbr  = (int*)  alloc((size_t)BN * KNN * sizeof(int));
  (void)ws_size; (void)in_sizes; (void)n_in; (void)out_size;

  norms_kernel<<<BN, 256, 0, stream>>>(x, sq);
  gram_topk_kernel<<<dim3(BATCH, 7), 128, 0, stream>>>(x, sq, nbr);
  gemm_h_kernel<<<dim3(BN / 128, FDIM / 256), 256, 0, stream>>>(x, W, h);
  edge_kernel<<<BN, 128, 0, stream>>>(h, a_src, a_dst, esrc, edst);
  gather_kernel<<<BN, 128, 0, stream>>>(h, esrc, edst, nbr, out);
}
